// GraphDocSelfAttention_15367392985625
// MI455X (gfx1250) — matrix operand order repaired and checked by reading, not compiled, not run on hardware
//
#include <hip/hip_runtime.h>

#define DEVI __device__ __forceinline__

typedef __bf16 bf16;
typedef __attribute__((ext_vector_type(16))) __bf16 v16bf;
typedef __attribute__((ext_vector_type(8)))  __bf16 v8bf;
typedef __attribute__((ext_vector_type(8)))  float  v8f;
typedef __attribute__((ext_vector_type(4)))  unsigned v4u;
typedef __attribute__((ext_vector_type(8)))  int      v8i;
typedef __attribute__((ext_vector_type(4)))  int      v4i;

namespace cfg {
constexpr int B = 2, S = 1024, J = 64, H = 12, DH = 64;
constexpr int D  = H * DH;     // 768
constexpr int M  = B * S;      // 2048
constexpr int N3 = 3 * D;      // 2304
constexpr float SCALE = 0.125f;   // 1/sqrt(DH)
constexpr float NEGV  = -1e8f;
}
using namespace cfg;

DEVI v8f zero8() {
  v8f c;
#pragma unroll
  for (int i = 0; i < 8; ++i) c[i] = 0.f;
  return c;
}

// ---------------------------------------------------------------- TDM helper
// 2D tile DMA global -> LDS via the Tensor Data Mover (D# per ISA ch.8).
// tile_x elements (data_size units) per row, tile_y rows, row stride in elems.
// data_size code: 0=1B 1=2B 2=4B 3=8B.  Caller must be wave-uniform.
DEVI void tdm_load_2d(void* lds_dst, const void* gsrc, unsigned tile_x,
                      unsigned tile_y, unsigned long long stride_elems,
                      unsigned dsz_code) {
  unsigned lds_addr = (unsigned)(unsigned long long)(uintptr_t)lds_dst;
  unsigned long long ga = (unsigned long long)(uintptr_t)gsrc;
  unsigned td0 = 1u << 20, td1 = 1u << 20;   // tiles are always in-bounds
  v4u g0;
  g0[0] = 1u;                                          // count=1, user mode
  g0[1] = lds_addr;                                    // lds_addr[31:0]
  g0[2] = (unsigned)(ga & 0xffffffffu);                // global_addr[31:0]
  g0[3] = (unsigned)((ga >> 32) & 0x1ffffffu) | (2u << 30);  // [56:32] | type=2
  v8i g1;
  g1[0] = (int)(dsz_code << 16);                       // wg_mask=0, data_size
  g1[1] = (int)((td0 & 0xffffu) << 16);                // tensor_dim0[15:0]
  g1[2] = (int)((td0 >> 16) | ((td1 & 0xffffu) << 16));
  g1[3] = (int)((td1 >> 16) | (tile_x << 16));         // tile_dim0
  g1[4] = (int)tile_y;                                 // tile_dim1 (tile_dim2=0)
  g1[5] = (int)(stride_elems & 0xffffffffu);           // dim0_stride[31:0]
  g1[6] = (int)((stride_elems >> 32) & 0xffffu);       // dim0_stride[47:32]
  g1[7] = 0;                                           // dim1_stride unused (2D)
  v4i g2, g3;
#pragma unroll
  for (int i = 0; i < 4; ++i) { g2[i] = 0; g3[i] = 0; }
  v8i g4;
#pragma unroll
  for (int i = 0; i < 8; ++i) g4[i] = 0;
  __builtin_amdgcn_tensor_load_to_lds(g0, g1, g2, g3, g4, 0);
}

// A-matrix fragment, 16x32 bf16, source stored row-major [row][k] with stride ldk.
DEVI v16bf fragA_rowmajor(const bf16* tile, int row0, int k0, int ldk, int lane) {
  int r  = row0 + (lane & 15);
  int kb = k0 + ((lane & 16) ? 16 : 0);
  const bf16* p = tile + (size_t)r * ldk + kb;
  v8bf lo = *(const v8bf*)p;
  v8bf hi = *(const v8bf*)(p + 8);
  v16bf out;
#pragma unroll
  for (int i = 0; i < 8; ++i) { out[i] = lo[i]; out[i + 8] = hi[i]; }
  return out;
}

// B-matrix fragment, 32x16 bf16 (KxN), source stored as [n][k] row-major.
DEVI v16bf fragB_nk(const bf16* tile, int n0, int k0, int ldk, int lane) {
  int n  = n0 + (lane & 15);
  int kb = k0 + ((lane & 16) ? 16 : 0);
  return *(const v16bf*)(tile + (size_t)n * ldk + kb);
}

DEVI v8f wmma_bf16(v16bf a, v16bf b, v8f c) {
  return __builtin_amdgcn_wmma_f32_16x16x32_bf16(false, a, false, b, (short)0, c,
                                                 false, false);
}

// ---------------------------------------------------------------- converts
__global__ void k_tobf16(const float* __restrict__ src, bf16* __restrict__ dst, int n) {
  for (int i = blockIdx.x * blockDim.x + threadIdx.x; i < n; i += gridDim.x * blockDim.x)
    dst[i] = (bf16)src[i];
}

// [B,S,D] fp32 -> [B,H,S,DH] bf16 (optionally scaled)
__global__ void k_split_heads(const float* __restrict__ src, bf16* __restrict__ dst,
                              float scale) {
  int total = B * H * S * DH;
  for (int idx = blockIdx.x * blockDim.x + threadIdx.x; idx < total;
       idx += gridDim.x * blockDim.x) {
    int dh = idx & (DH - 1);
    int rest = idx >> 6;
    int i = rest & (S - 1); rest >>= 10;
    int h = rest % H;
    int b = rest / H;
    dst[idx] = (bf16)(src[((size_t)b * S + i) * D + h * DH + dh] * scale);
  }
}

// ---------------------------------------------------------------- qkv GEMM
// C[2048,2304] = hs_bf16 @ qkv_w_bf16^T ; epilogue routes q/k/v with biases.
__global__ void k_gemm_qkv(const bf16* __restrict__ Ab, const bf16* __restrict__ Wb,
                           const float* __restrict__ q_bias,
                           const float* __restrict__ v_bias,
                           bf16* __restrict__ qflat, bf16* __restrict__ qh,
                           bf16* __restrict__ kh, bf16* __restrict__ vh) {
  __shared__ alignas(32) bf16 At[128 * 32];
  __shared__ alignas(32) bf16 Bt[64 * 32];
  int t = threadIdx.x, lane = t & 31, w = t >> 5;
  int n0 = blockIdx.x * 64, m0 = blockIdx.y * 128;
  v8f c[4];
#pragma unroll
  for (int i = 0; i < 4; ++i) c[i] = zero8();

  for (int k0 = 0; k0 < D; k0 += 32) {
    __syncthreads();                       // previous tiles fully consumed
    if (w == 0) {                          // wave 0 drives the TDM
      tdm_load_2d(At, Ab + (size_t)m0 * D + k0, 32, 128, D, 1);
      tdm_load_2d(Bt, Wb + (size_t)n0 * D + k0, 32, 64, D, 1);
      __builtin_amdgcn_s_wait_tensorcnt(0);
    }
    __syncthreads();                       // LDS tiles visible to all waves
    v16bf a = fragA_rowmajor(At, w * 16, 0, 32, lane);
#pragma unroll
    for (int nt = 0; nt < 4; ++nt)
      c[nt] = wmma_bf16(a, fragB_nk(Bt, nt * 16, 0, 32, lane), c[nt]);
  }

  int mrow = m0 + w * 16 + ((lane >> 4) << 3);
  int ncol = n0 + (lane & 15);
#pragma unroll
  for (int nt = 0; nt < 4; ++nt) {
    int n = ncol + nt * 16;
#pragma unroll
    for (int i2 = 0; i2 < 8; ++i2) {
      int m = mrow + i2;
      int b = m >> 10, ii = m & (S - 1);
      float val = c[nt][i2];
      if (n < D) {                                   // q
        float qv = val + q_bias[n];
        qflat[(size_t)m * D + n] = (bf16)qv;
        int h = n >> 6, dh = n & 63;
        qh[(((size_t)(b * H + h)) * S + ii) * DH + dh] = (bf16)(qv * SCALE);
      } else if (n < 2 * D) {                        // k
        int nn = n - D;
        int h = nn >> 6, dh = nn & 63;
        kh[(((size_t)(b * H + h)) * S + ii) * DH + dh] = (bf16)val;
      } else {                                       // v
        int nn = n - 2 * D;
        float vv = val + v_bias[nn];
        int h = nn >> 6, dh = nn & 63;
        vh[(((size_t)(b * H + h)) * S + ii) * DH + dh] = (bf16)vv;
      }
    }
  }
}

// ---------------------------------------------------------------- rbq GEMM
__global__ void k_gemm_rbq(const bf16* __restrict__ Ab, const bf16* __restrict__ Wb,
                           const float* __restrict__ rbq_b, bf16* __restrict__ qbh) {
  __shared__ alignas(32) bf16 At[128 * 32];
  __shared__ alignas(32) bf16 Bt[64 * 32];
  int t = threadIdx.x, lane = t & 31, w = t >> 5;
  int n0 = blockIdx.x * 64, m0 = blockIdx.y * 128;
  v8f c[4];
#pragma unroll
  for (int i = 0; i < 4; ++i) c[i] = zero8();

  for (int k0 = 0; k0 < D; k0 += 32) {
    __syncthreads();
    if (w == 0) {
      tdm_load_2d(At, Ab + (size_t)m0 * D + k0, 32, 128, D, 1);
      tdm_load_2d(Bt, Wb + (size_t)n0 * D + k0, 32, 64, D, 1);
      __builtin_amdgcn_s_wait_tensorcnt(0);
    }
    __syncthreads();
    v16bf a = fragA_rowmajor(At, w * 16, 0, 32, lane);
#pragma unroll
    for (int nt = 0; nt < 4; ++nt)
      c[nt] = wmma_bf16(a, fragB_nk(Bt, nt * 16, 0, 32, lane), c[nt]);
  }

  int mrow = m0 + w * 16 + ((lane >> 4) << 3);
  int ncol = n0 + (lane & 15);
#pragma unroll
  for (int nt = 0; nt < 4; ++nt) {
    int n = ncol + nt * 16;
    int h = n >> 6, dh = n & 63;
    float bias = rbq_b[n];
#pragma unroll
    for (int i2 = 0; i2 < 8; ++i2) {
      int m = mrow + i2;
      int b = m >> 10, ii = m & (S - 1);
      qbh[(((size_t)(b * H + h)) * S + ii) * DH + dh] =
          (bf16)((c[nt][i2] + bias) * SCALE);
    }
  }
}

// ---------------------------------------------------------------- scores
// per (b,h): scores = (Q*scale)@K^T + (LQ*scale)@LK^T   (fp32 out)
__global__ void k_scores(const bf16* __restrict__ qh, const bf16* __restrict__ kh,
                         const bf16* __restrict__ lqh, const bf16* __restrict__ lkh,
                         float* __restrict__ scores) {
  int n0 = blockIdx.x * 128, m0 = blockIdx.y * 128, bh = blockIdx.z;
  const bf16* Q  = qh  + (size_t)bh * S * DH;
  const bf16* K  = kh  + (size_t)bh * S * DH;
  const bf16* LQ = lqh + (size_t)bh * S * DH;
  const bf16* LK = lkh + (size_t)bh * S * DH;
  __shared__ alignas(32) bf16 Kt[128 * 64];
  __shared__ alignas(32) bf16 LKt[128 * 64];
  int t = threadIdx.x, lane = t & 31, w = t >> 5;

  // K/LK strips are contiguous (row stride == tile width): 1D TDM copies.
  if (w == 0) {
    tdm_load_2d(Kt,  K  + (size_t)n0 * DH, 128 * DH, 1, 128 * DH, 1);
    tdm_load_2d(LKt, LK + (size_t)n0 * DH, 128 * DH, 1, 128 * DH, 1);
    __builtin_amdgcn_s_wait_tensorcnt(0);
  }
  __syncthreads();

  int arow = m0 + w * 16;
  v16bf aq0 = fragA_rowmajor(Q,  arow, 0,  DH, lane);
  v16bf aq1 = fragA_rowmajor(Q,  arow, 32, DH, lane);
  v16bf al0 = fragA_rowmajor(LQ, arow, 0,  DH, lane);
  v16bf al1 = fragA_rowmajor(LQ, arow, 32, DH, lane);

#pragma unroll
  for (int nt = 0; nt < 8; ++nt) {
    v8f c = zero8();
    c = wmma_bf16(aq0, fragB_nk(Kt,  nt * 16, 0,  DH, lane), c);
    c = wmma_bf16(aq1, fragB_nk(Kt,  nt * 16, 32, DH, lane), c);
    c = wmma_bf16(al0, fragB_nk(LKt, nt * 16, 0,  DH, lane), c);
    c = wmma_bf16(al1, fragB_nk(LKt, nt * 16, 32, DH, lane), c);
    int n = n0 + nt * 16 + (lane & 15);
    int mloc = arow + ((lane >> 4) << 3);
#pragma unroll
    for (int i2 = 0; i2 < 8; ++i2)
      scores[((size_t)bh * S + mloc + i2) * S + n] = c[i2];
  }
}

// ---------------------------------------------------------------- bbox + scatter
__global__ void k_bbox(const bf16* __restrict__ qbh, const float* __restrict__ rbe,
                       const int* __restrict__ rbi, float* __restrict__ scores) {
  int bi = blockIdx.x;
  int b = bi >> 10, i = bi & (S - 1);
  __shared__ float qloc[H * DH];
  int t = threadIdx.x;
  for (int idx = t; idx < H * DH; idx += 256) {
    int h = idx >> 6, dh = idx & 63;
    qloc[idx] = (float)qbh[(((size_t)(b * H + h)) * S + i) * DH + dh];
  }
  __syncthreads();

  const float* rb  = rbe + (size_t)bi * J * D;
  const int* ridx  = rbi + (size_t)bi * J;
  for (int pi = t; pi < J * H; pi += 256) {
    int h = pi % H;
    int j = pi / H;
    const float* p = rb + (size_t)j * D + h * DH;
    const float* q = qloc + h * DH;
    __builtin_prefetch(p, 0, 0);
    float acc = 0.f;
#pragma unroll
    for (int d0 = 0; d0 < DH; d0 += 4) {
      float4 x = *(const float4*)(p + d0);
      acc += x.x * q[d0] + x.y * q[d0 + 1] + x.z * q[d0 + 2] + x.w * q[d0 + 3];
    }
    int kj = ridx[j];
    atomicAdd(scores + (((size_t)(b * H + h)) * S + i) * S + kj, acc);
  }
}

// ---------------------------------------------------------------- softmax
__global__ void k_softmax(const float* __restrict__ scores,
                          const int* __restrict__ local_mask,
                          const int* __restrict__ attn_mask,
                          bf16* __restrict__ probs) {
  int i = blockIdx.x, h = blockIdx.y, b = blockIdx.z;
  size_t rowo = ((size_t)(b * H + h) * S + i) * S;
  const float* row = scores + rowo;
  const int* lm = local_mask + ((size_t)(b * S + i)) * S;
  const int* am = attn_mask + (size_t)b * S;
  int t = threadIdx.x, lane = t & 31, w = t >> 5;

  float v[4];
  float mx = -3.4e38f;
#pragma unroll
  for (int u = 0; u < 4; ++u) {
    int j = t * 4 + u;
    float x = row[j];
    if (lm[j] == 0 || am[j] != 0) x = NEGV;
    v[u] = x;
    mx = fmaxf(mx, x);
  }
#pragma unroll
  for (int o = 16; o > 0; o >>= 1) mx = fmaxf(mx, __shfl_xor(mx, o, 32));
  __shared__ float redm[8];
  __shared__ float reds[8];
  if (lane == 0) redm[w] = mx;
  __syncthreads();
#pragma unroll
  for (int w2 = 0; w2 < 8; ++w2) mx = fmaxf(mx, redm[w2]);

  float e[4], s = 0.f;
#pragma unroll
  for (int u = 0; u < 4; ++u) { e[u] = __expf(v[u] - mx); s += e[u]; }
#pragma unroll
  for (int o = 16; o > 0; o >>= 1) s += __shfl_xor(s, o, 32);
  if (lane == 0) reds[w] = s;
  __syncthreads();
  s = 0.f;
#pragma unroll
  for (int w2 = 0; w2 < 8; ++w2) s += reds[w2];
  float inv = 1.f / s;

  bf16* prow = probs + rowo;
#pragma unroll
  for (int u = 0; u < 4; ++u) prow[t * 4 + u] = (bf16)(e[u] * inv);
}

// ---------------------------------------------------------------- PV GEMM
// per (b,h): ctx[i,d] = probs[i,:] @ V[:,d]  -> out[b,i,h*64+d] fp32
__global__ void k_pv(const bf16* __restrict__ probs, const bf16* __restrict__ vh,
                     float* __restrict__ out) {
  int m0 = blockIdx.y * 128, bh = blockIdx.z;
  int b = bh / H, h = bh % H;
  const bf16* P = probs + (size_t)bh * S * S;
  const bf16* V = vh + (size_t)bh * S * DH;
  __shared__ alignas(32) bf16 VT[DH * 32];   // [d][j] , ldk = 32
  int t = threadIdx.x, lane = t & 31, w = t >> 5;
  v8f c[4];
#pragma unroll
  for (int i = 0; i < 4; ++i) c[i] = zero8();
  int arow = m0 + w * 16;

  for (int j0 = 0; j0 < S; j0 += 32) {
    __syncthreads();
#pragma unroll
    for (int it = 0; it < 4; ++it) {      // transpose 32x64 V slab into LDS
      int idx = t + it * 256;             // 0..1023 dword slots
      int jj = idx >> 5;
      int dp = (idx & 31) * 2;
      const bf16* vp = V + (size_t)(j0 + jj) * DH + dp;
      VT[(size_t)dp * 32 + jj]       = vp[0];
      VT[(size_t)(dp + 1) * 32 + jj] = vp[1];
    }
    __syncthreads();
    v16bf a = fragA_rowmajor(P, arow, j0, S, lane);
#pragma unroll
    for (int nt = 0; nt < 4; ++nt)
      c[nt] = wmma_bf16(a, fragB_nk(VT, nt * 16, 0, 32, lane), c[nt]);
  }

  int mloc = arow + ((lane >> 4) << 3);
  int ncol = lane & 15;
#pragma unroll
  for (int nt = 0; nt < 4; ++nt) {
    int n = nt * 16 + ncol;
#pragma unroll
    for (int i2 = 0; i2 < 8; ++i2) {
      int m = mloc + i2;
      out[((size_t)b * S + m) * D + h * DH + n] = c[nt][i2];
    }
  }
}

// ---------------------------------------------------------------- launch
extern "C" void kernel_launch(void* const* d_in, const int* in_sizes, int n_in,
                              void* d_out, int out_size, void* d_ws, size_t ws_size,
                              hipStream_t stream) {
  const float* hs     = (const float*)d_in[0];
  const float* lq     = (const float*)d_in[1];
  const float* lk     = (const float*)d_in[2];
  const float* rbe    = (const float*)d_in[3];
  const int*   rbi    = (const int*)d_in[4];
  const int*   am     = (const int*)d_in[5];
  const int*   lm     = (const int*)d_in[6];
  const float* qkv_w  = (const float*)d_in[7];
  const float* q_bias = (const float*)d_in[8];
  const float* v_bias = (const float*)d_in[9];
  const float* rbq_w  = (const float*)d_in[10];
  const float* rbq_b  = (const float*)d_in[11];
  float* out = (float*)d_out;

  char* p = (char*)d_ws;
  auto alloc = [&](size_t bytes) {
    char* r = p;
    p += (bytes + 255) & ~(size_t)255;
    return r;
  };
  bf16* hsb    = (bf16*)alloc((size_t)M * D * 2);
  bf16* wqkvb  = (bf16*)alloc((size_t)N3 * D * 2);
  bf16* wrbqb  = (bf16*)alloc((size_t)D * D * 2);
  bf16* lqh    = (bf16*)alloc((size_t)B * H * S * DH * 2);
  bf16* lkh    = (bf16*)alloc((size_t)B * H * S * DH * 2);
  bf16* qh     = (bf16*)alloc((size_t)B * H * S * DH * 2);
  bf16* kh     = (bf16*)alloc((size_t)B * H * S * DH * 2);
  bf16* vh     = (bf16*)alloc((size_t)B * H * S * DH * 2);
  bf16* qflat  = (bf16*)alloc((size_t)M * D * 2);
  bf16* qbh    = (bf16*)alloc((size_t)B * H * S * DH * 2);
  float* scores = (float*)alloc((size_t)B * H * S * S * 4);
  bf16*  probs  = (bf16*)alloc((size_t)B * H * S * S * 2);

  k_tobf16<<<2048, 256, 0, stream>>>(hs, hsb, M * D);
  k_tobf16<<<2048, 256, 0, stream>>>(qkv_w, wqkvb, N3 * D);
  k_tobf16<<<2048, 256, 0, stream>>>(rbq_w, wrbqb, D * D);
  k_split_heads<<<2048, 256, 0, stream>>>(lq, lqh, SCALE);
  k_split_heads<<<2048, 256, 0, stream>>>(lk, lkh, 1.0f);

  k_gemm_qkv<<<dim3(N3 / 64, M / 128), 256, 0, stream>>>(hsb, wqkvb, q_bias, v_bias,
                                                         qflat, qh, kh, vh);
  k_gemm_rbq<<<dim3(D / 64, M / 128), 256, 0, stream>>>(qflat, wrbqb, rbq_b, qbh);

  k_scores<<<dim3(S / 128, S / 128, B * H), 256, 0, stream>>>(qh, kh, lqh, lkh, scores);
  k_bbox<<<B * S, 256, 0, stream>>>(qbh, rbe, rbi, scores);
  k_softmax<<<dim3(S, H, B), 256, 0, stream>>>(scores, lm, am, probs);
  k_pv<<<dim3(1, S / 128, B * H), 256, 0, stream>>>(probs, vh, out);
}
